// VoronoiProjection_50225347559700
// MI455X (gfx1250) — matrix operand order repaired and checked by reading, not compiled, not run on hardware
//
#include <hip/hip_runtime.h>

typedef float  v8f   __attribute__((ext_vector_type(8)));
typedef __bf16 v16bf __attribute__((ext_vector_type(16)));
typedef __bf16 v8bf  __attribute__((ext_vector_type(8)));
typedef __bf16 v4bf  __attribute__((ext_vector_type(4)));
typedef unsigned short u16;
typedef int v4i __attribute__((vector_size(16)));   // matches builtin's pointee type

#define NI 16
#define NK 1024
#define NB 4096
#define ND 64

#define WAVES 4
#define ROWS_PER_WAVE 16
#define ROWS_PER_BLOCK (WAVES * ROWS_PER_WAVE)   // 64
#define NTILES (NK / ROWS_PER_WAVE)              // 64
#define FSTRIDE 72   // bf16 elems per LDS row (144B: 16B-aligned, 16-lane bank-perfect)

// ---------------- CDNA5 async global->LDS helpers ----------------------------
#if defined(__has_builtin)
#if __has_builtin(__builtin_amdgcn_global_load_async_to_lds_b128)
#define HAVE_ASYNC_B128 1
#endif
#if __has_builtin(__builtin_amdgcn_s_wait_asynccnt)
#define HAVE_WAIT_ASYNC 1
#endif
#endif

__device__ __forceinline__ void async_copy16(const void* g, void* l) {
#ifdef HAVE_ASYNC_B128
    __builtin_amdgcn_global_load_async_to_lds_b128(
        (__attribute__((address_space(1))) v4i*)g,
        (__attribute__((address_space(3))) v4i*)l, 0, 0);
#else
    // VDST = LDS byte address VGPR, VADDR = 64-bit global address (GV mode)
    unsigned int lo = (unsigned int)(unsigned long long)l;
    asm volatile("global_load_async_to_lds_b128 %0, %1, off"
                 :: "v"(lo), "v"(g) : "memory");
#endif
}

__device__ __forceinline__ void wait_async_le2() {
#ifdef HAVE_WAIT_ASYNC
    __builtin_amdgcn_s_wait_asynccnt(2);
#else
    asm volatile("s_wait_asynccnt 0x2" ::: "memory");
#endif
}
__device__ __forceinline__ void wait_async_le0() {
#ifdef HAVE_WAIT_ASYNC
    __builtin_amdgcn_s_wait_asynccnt(0);
#else
    asm volatile("s_wait_asynccnt 0x0" ::: "memory");
#endif
}

__device__ __forceinline__ u16 bf16_bits(float v) {
    union { __bf16 b; u16 u; } c;
    c.b = (__bf16)v;
    return c.u;
}
__device__ __forceinline__ float bf16_round(float v) {
    return (float)(__bf16)v;
}

// ---- pass 1 (fast path): fp32 ||F||^2 + split F into bf16 hi/lo planes ------
__global__ __launch_bounds__(256) void VoronoiSplit_kernel(const float* __restrict__ F,
                                                           float* __restrict__ fsq,
                                                           u16* __restrict__ Fhi,
                                                           u16* __restrict__ Flo) {
    int r = blockIdx.x * blockDim.x + threadIdx.x;        // flat i*NK + k
    if (r >= NI * NK) return;
    const float4* row = (const float4*)(F + (size_t)r * ND);
    u16* ph = Fhi + (size_t)r * ND;
    u16* pl = Flo + (size_t)r * ND;
    float s = 0.f;
#pragma unroll
    for (int t = 0; t < ND / 4; ++t) {
        float4 v = row[t];
        float a[4] = { v.x, v.y, v.z, v.w };
        ushort4 h4, l4;
        u16 hq[4], lq[4];
#pragma unroll
        for (int q = 0; q < 4; ++q) {
            s += a[q] * a[q];
            hq[q] = bf16_bits(a[q]);
            lq[q] = bf16_bits(a[q] - bf16_round(a[q]));
        }
        h4.x = hq[0]; h4.y = hq[1]; h4.z = hq[2]; h4.w = hq[3];
        l4.x = lq[0]; l4.y = lq[1]; l4.z = lq[2]; l4.w = lq[3];
        *(ushort4*)(ph + t * 4) = h4;
        *(ushort4*)(pl + t * 4) = l4;
    }
    fsq[r] = s;
}

// ---- pass 2 (fast path): async-staged WMMA cross-term, argmin, gather -------
__global__ __launch_bounds__(128) void VoronoiProjAsync_kernel(const float* __restrict__ F,
                                                               const float* __restrict__ x,
                                                               const float* __restrict__ fsq,
                                                               const u16* __restrict__ Fhi,
                                                               const u16* __restrict__ Flo,
                                                               float* __restrict__ out) {
    __shared__ __attribute__((aligned(16))) __bf16 fbh[2][ROWS_PER_WAVE * FSTRIDE];
    __shared__ __attribute__((aligned(16))) __bf16 fbl[2][ROWS_PER_WAVE * FSTRIDE];
    __shared__ int sidx[ROWS_PER_BLOCK];

    const int i    = blockIdx.y;
    const int b0   = blockIdx.x * ROWS_PER_BLOCK;
    const int tid  = threadIdx.x;
    const int wave = tid >> 5;
    const int lane = tid & 31;
    const int laneN = lane & 15;
    const int half  = lane >> 4;

    // A fragments: bf16 hi/lo split of this wave's 16 x-rows, resident in VGPRs.
    // A k order used here (16x32, same as B): lanes 0-15 -> M=lane, K=0..15;
    //                           lanes 16-31 -> M=lane-16, K=16..31.
    const int xrow = b0 + wave * ROWS_PER_WAVE + laneN;
    const float* xr = x + ((size_t)i * NB + xrow) * ND;
    v16bf ah[2], al[2];
#pragma unroll
    for (int c = 0; c < 2; ++c) {
#pragma unroll
        for (int s = 0; s < 2; ++s) {
            const float* p = xr + c * 32 + half * 16 + s * 8;
#pragma unroll
            for (int e = 0; e < 8; ++e) {
                float  v = p[e];
                __bf16 h = (__bf16)v;
                ah[c][s * 8 + e] = h;
                al[c][s * 8 + e] = (__bf16)(v - (float)h);
            }
        }
    }

    float bestv[8];
    int   besti[8];
#pragma unroll
    for (int r = 0; r < 8; ++r) { bestv[r] = 3.0e38f; besti[r] = 0; }

    const float* Fbase = F   + (size_t)i * NK * ND;
    const float* fsqb  = fsq + (size_t)i * NK;
    const __bf16* GH = (const __bf16*)Fhi + (size_t)i * NK * ND;
    const __bf16* GL = (const __bf16*)Flo + (size_t)i * NK * ND;

    // each thread owns one 16B chunk of the 16x64 bf16 tile (128 x 16B = 2KB)
    const int n8   = tid >> 3;            // tile row
    const int c8   = tid & 7;             // 16B chunk within row
    const int loff = n8 * FSTRIDE + c8 * 8;
    const size_t goff = (size_t)tid * 8;  // bf16 elements into the tile

    // prime double buffer with tile 0
    async_copy16(GH + goff, &fbh[0][loff]);
    async_copy16(GL + goff, &fbl[0][loff]);

    for (int t = 0; t < NTILES; ++t) {
        const int kb = t * ROWS_PER_WAVE;
        if (t + 1 < NTILES) {             // overlap next tile's fetch with compute
            const int nb = (t + 1) & 1;
            const size_t g = (size_t)(kb + ROWS_PER_WAVE) * ND + goff;
            async_copy16(GH + g, &fbh[nb][loff]);
            async_copy16(GL + g, &fbl[nb][loff]);
            wait_async_le2();             // tile t's 2 async ops complete
        } else {
            wait_async_le0();
        }
        __syncthreads();

        const int cb = t & 1;
        // B k order used here (32x16, same as A): per lane 16 consecutive K at N=lane&15,
        // K-base = 16*(lane>>4) -> 16 contiguous bf16 of one F row in LDS.
        v8f acc = {};
#pragma unroll
        for (int c = 0; c < 2; ++c) {
            const int off = laneN * FSTRIDE + c * 32 + half * 16;
            union { v16bf v; v8bf h[2]; } bh, bl;
            bh.h[0] = *(const v8bf*)(&fbh[cb][off]);
            bh.h[1] = *(const v8bf*)(&fbh[cb][off + 8]);
            bl.h[0] = *(const v8bf*)(&fbl[cb][off]);
            bl.h[1] = *(const v8bf*)(&fbl[cb][off + 8]);
            // cross ~= xh*fh + xh*fl + xl*fh  (bf16x2 split, f32 accumulate)
            acc = __builtin_amdgcn_wmma_f32_16x16x32_bf16(false, ah[c], false, bh.v,
                                                          (short)0, acc, false, false);
            acc = __builtin_amdgcn_wmma_f32_16x16x32_bf16(false, ah[c], false, bl.v,
                                                          (short)0, acc, false, false);
            acc = __builtin_amdgcn_wmma_f32_16x16x32_bf16(false, al[c], false, bh.v,
                                                          (short)0, acc, false, false);
        }

        const float fs  = fsqb[kb + laneN];
        const int   col = kb + laneN;
#pragma unroll
        for (int r = 0; r < 8; ++r) {
            float sc = fs - 2.0f * acc[r];
            if (sc < bestv[r]) { bestv[r] = sc; besti[r] = col; }   // strict: first min
        }
        __syncthreads();                  // all reads of buf[cb] done before re-stage
    }

#pragma unroll
    for (int m = 1; m < 16; m <<= 1) {
#pragma unroll
        for (int r = 0; r < 8; ++r) {
            float ov = __shfl_xor(bestv[r], m, 32);
            int   oi = __shfl_xor(besti[r], m, 32);
            if (ov < bestv[r] || (ov == bestv[r] && oi < besti[r])) {
                bestv[r] = ov; besti[r] = oi;
            }
        }
    }
    if (laneN == 0) {
#pragma unroll
        for (int r = 0; r < 8; ++r)
            sidx[wave * 16 + half * 8 + r] = besti[r];
    }
    __syncthreads();

#pragma unroll
    for (int m = 0; m < ROWS_PER_WAVE; ++m) {
        int k = sidx[wave * 16 + m];
        const float2 v = *(const float2*)(Fbase + (size_t)k * ND + 2 * lane);
        *(float2*)(out + ((size_t)i * NB + b0 + wave * ROWS_PER_WAVE + m) * ND + 2 * lane) = v;
    }
}

// =================== fallback path (small workspace): round-1 kernels ========
__global__ __launch_bounds__(256) void VoronoiFsq_kernel(const float* __restrict__ F,
                                                         float* __restrict__ fsq) {
    int idx = blockIdx.x * blockDim.x + threadIdx.x;
    if (idx >= NI * NK) return;
    const float4* row = (const float4*)(F + (size_t)idx * ND);
    float s = 0.f;
#pragma unroll
    for (int t = 0; t < ND / 4; ++t) {
        float4 v = row[t];
        s += v.x * v.x + v.y * v.y + v.z * v.z + v.w * v.w;
    }
    fsq[idx] = s;
}

__global__ __launch_bounds__(128) void VoronoiProjConv_kernel(const float* __restrict__ F,
                                                              const float* __restrict__ x,
                                                              const float* __restrict__ fsq,
                                                              float* __restrict__ out) {
    __shared__ __attribute__((aligned(16))) __bf16 fhi[ROWS_PER_WAVE * FSTRIDE];
    __shared__ __attribute__((aligned(16))) __bf16 flo[ROWS_PER_WAVE * FSTRIDE];
    __shared__ int sidx[ROWS_PER_BLOCK];

    const int i    = blockIdx.y;
    const int b0   = blockIdx.x * ROWS_PER_BLOCK;
    const int tid  = threadIdx.x;
    const int wave = tid >> 5;
    const int lane = tid & 31;
    const int laneN = lane & 15;
    const int half  = lane >> 4;

    const int xrow = b0 + wave * ROWS_PER_WAVE + laneN;
    const float* xr = x + ((size_t)i * NB + xrow) * ND;
    v16bf ah[2], al[2];
#pragma unroll
    for (int c = 0; c < 2; ++c) {
#pragma unroll
        for (int s = 0; s < 2; ++s) {
            const float* p = xr + c * 32 + half * 16 + s * 8;
#pragma unroll
            for (int e = 0; e < 8; ++e) {
                float  v = p[e];
                __bf16 h = (__bf16)v;
                ah[c][s * 8 + e] = h;
                al[c][s * 8 + e] = (__bf16)(v - (float)h);
            }
        }
    }

    float bestv[8];
    int   besti[8];
#pragma unroll
    for (int r = 0; r < 8; ++r) { bestv[r] = 3.0e38f; besti[r] = 0; }

    const float* Fbase = F   + (size_t)i * NK * ND;
    const float* fsqb  = fsq + (size_t)i * NK;

    for (int kb = 0; kb < NK; kb += ROWS_PER_WAVE) {
        const float* src = Fbase + (size_t)kb * ND;
#pragma unroll
        for (int p = 0; p < 2; ++p) {
            int f = (tid + p * 128) * 4;
            float4 v = *(const float4*)(src + f);
            int n = f >> 6, d = f & 63;
            v4bf h4, l4;
            float vv[4] = { v.x, v.y, v.z, v.w };
#pragma unroll
            for (int q = 0; q < 4; ++q) {
                __bf16 h = (__bf16)vv[q];
                h4[q] = h;
                l4[q] = (__bf16)(vv[q] - (float)h);
            }
            *(v4bf*)(&fhi[n * FSTRIDE + d]) = h4;
            *(v4bf*)(&flo[n * FSTRIDE + d]) = l4;
        }
        if (kb + ROWS_PER_WAVE < NK)
            __builtin_prefetch(src + ROWS_PER_WAVE * ND + tid * 8, 0, 0);
        __syncthreads();

        v8f acc = {};
#pragma unroll
        for (int c = 0; c < 2; ++c) {
            const int off = laneN * FSTRIDE + c * 32 + half * 16;
            union { v16bf v; v8bf h[2]; } bh, bl;
            bh.h[0] = *(const v8bf*)(&fhi[off]);
            bh.h[1] = *(const v8bf*)(&fhi[off + 8]);
            bl.h[0] = *(const v8bf*)(&flo[off]);
            bl.h[1] = *(const v8bf*)(&flo[off + 8]);
            acc = __builtin_amdgcn_wmma_f32_16x16x32_bf16(false, ah[c], false, bh.v,
                                                          (short)0, acc, false, false);
            acc = __builtin_amdgcn_wmma_f32_16x16x32_bf16(false, ah[c], false, bl.v,
                                                          (short)0, acc, false, false);
            acc = __builtin_amdgcn_wmma_f32_16x16x32_bf16(false, al[c], false, bh.v,
                                                          (short)0, acc, false, false);
        }

        const float fs  = fsqb[kb + laneN];
        const int   col = kb + laneN;
#pragma unroll
        for (int r = 0; r < 8; ++r) {
            float sc = fs - 2.0f * acc[r];
            if (sc < bestv[r]) { bestv[r] = sc; besti[r] = col; }
        }
        __syncthreads();
    }

#pragma unroll
    for (int m = 1; m < 16; m <<= 1) {
#pragma unroll
        for (int r = 0; r < 8; ++r) {
            float ov = __shfl_xor(bestv[r], m, 32);
            int   oi = __shfl_xor(besti[r], m, 32);
            if (ov < bestv[r] || (ov == bestv[r] && oi < besti[r])) {
                bestv[r] = ov; besti[r] = oi;
            }
        }
    }
    if (laneN == 0) {
#pragma unroll
        for (int r = 0; r < 8; ++r)
            sidx[wave * 16 + half * 8 + r] = besti[r];
    }
    __syncthreads();

#pragma unroll
    for (int m = 0; m < ROWS_PER_WAVE; ++m) {
        int k = sidx[wave * 16 + m];
        const float2 v = *(const float2*)(Fbase + (size_t)k * ND + 2 * lane);
        *(float2*)(out + ((size_t)i * NB + b0 + wave * ROWS_PER_WAVE + m) * ND + 2 * lane) = v;
    }
}

extern "C" void kernel_launch(void* const* d_in, const int* in_sizes, int n_in,
                              void* d_out, int out_size, void* d_ws, size_t ws_size,
                              hipStream_t stream) {
    const float* F = (const float*)d_in[0];   // [I, K, D] fp32
    const float* x = (const float*)d_in[1];   // [I, B, D] fp32
    float* out = (float*)d_out;               // [I, B, D] fp32

    const size_t fsq_bytes   = (size_t)NI * NK * sizeof(float);          // 64 KB
    const size_t plane_bytes = (size_t)NI * NK * ND * sizeof(u16);       // 2 MB
    const size_t need = fsq_bytes + 2 * plane_bytes;

    float* fsq = (float*)d_ws;
    dim3 grid(NB / ROWS_PER_BLOCK, NI);       // (64, 16)

    if (ws_size >= need) {
        u16* Fhi = (u16*)((char*)d_ws + fsq_bytes);
        u16* Flo = (u16*)((char*)d_ws + fsq_bytes + plane_bytes);
        VoronoiSplit_kernel<<<(NI * NK + 255) / 256, 256, 0, stream>>>(F, fsq, Fhi, Flo);
        VoronoiProjAsync_kernel<<<grid, WAVES * 32, 0, stream>>>(F, x, fsq, Fhi, Flo, out);
    } else {
        VoronoiFsq_kernel<<<(NI * NK + 255) / 256, 256, 0, stream>>>(F, fsq);
        VoronoiProjConv_kernel<<<grid, WAVES * 32, 0, stream>>>(F, x, fsq, out);
    }
}
